// ConvLRUBlock_326417514520
// MI455X (gfx1250) — matrix operand order repaired and checked by reading, compiled, not run on hardware
//
#include <hip/hip_runtime.h>
#include <math.h>

// ---------------- problem constants ----------------
constexpr int B_  = 2;
constexpr int L_  = 16;
constexpr int BL_ = B_ * L_;          // 32
constexpr int C_  = 64;
constexpr int H_  = 64;
constexpr int W_  = 128;
constexpr int HW_ = H_ * W_;          // 8192
constexpr int R_  = 32;
constexpr int MH_ = 32;
constexpr int M1_ = 8;
constexpr int M2_ = 8;
constexpr int HM_ = 16;               // 16 retained h-modes (0..7, 56..63)

#define PI_F 3.14159265358979323846f

// ---------------- WMMA bf16 16x16x32 helpers ----------------
typedef float  v8f   __attribute__((ext_vector_type(8)));
typedef __bf16 v16bf __attribute__((ext_vector_type(16)));
typedef unsigned short v8us  __attribute__((ext_vector_type(8)));
typedef unsigned short v16us __attribute__((ext_vector_type(16)));

__device__ __forceinline__ v8f v8f_zero() {
  v8f v;
#pragma unroll
  for (int i = 0; i < 8; ++i) v[i] = 0.0f;
  return v;
}

// float -> bf16 (round to nearest even), bit-level
__device__ __forceinline__ unsigned short f2bf(float f) {
  unsigned int u = __float_as_uint(f);
  u += 0x7fffu + ((u >> 16) & 1u);
  return (unsigned short)(u >> 16);
}

#if defined(__gfx1250__) && __has_builtin(__builtin_amdgcn_wmma_f32_16x16x32_bf16)
#define HAS_WMMA_BF16 1
#endif

__device__ __forceinline__ v8f wmma_bf(v16bf a, v16bf b, v8f c) {
#ifdef HAS_WMMA_BF16
  // D = A(16x32) * B(32x16) + C  -> v_wmma_f32_16x16x32_bf16
  return __builtin_amdgcn_wmma_f32_16x16x32_bf16(false, a, false, b, (short)0, c,
                                                 false, false);
#else
  (void)a; (void)b;  // compile-only fallback (host pass)
  return c;
#endif
}

// A fragment: A row-major [16][lda] (bf16 stored as ushort).
// lane m = lane&15; per-lane K: {kb..kb+7, kb+16..kb+23}, kb = 8*(lane>=16)
__device__ __forceinline__ v16bf load_a_bf(const unsigned short* A, int lda, int kt,
                                           int lane) {
  int m = lane & 15;
  int base = m * lda + kt * 32 + ((lane >> 4) << 3);
  v8us lo = *(const v8us*)(A + base);
  v8us hi = *(const v8us*)(A + base + 16);
  v16us c = __builtin_shufflevector(lo, hi, 0, 1, 2, 3, 4, 5, 6, 7, 8, 9, 10, 11, 12,
                                    13, 14, 15);
  return __builtin_bit_cast(v16bf, c);
}
// B fragment from transposed storage Bt[N][ldb] (B[k][n] = Bt[n*ldb+k]).
// lane n = lane&15; per-lane K: {kb..kb+7, kb+16..kb+23}, kb = 8*(lane>=16)
__device__ __forceinline__ v16bf load_bt_bf(const unsigned short* Bt, int ldb, int kt,
                                            int lane) {
  int n = lane & 15;
  int base = n * ldb + kt * 32 + ((lane >> 4) << 3);
  v8us lo = *(const v8us*)(Bt + base);
  v8us hi = *(const v8us*)(Bt + base + 16);
  v16us c = __builtin_shufflevector(lo, hi, 0, 1, 2, 3, 4, 5, 6, 7, 8, 9, 10, 11, 12,
                                    13, 14, 15);
  return __builtin_bit_cast(v16bf, c);
}

// ---------------- K0: spatial mean -> ctx ----------------
__global__ void k_ctx(const float* __restrict__ x, float* __restrict__ ctx) {
  int blc = blockIdx.x;
  const float* p = x + (size_t)blc * HW_;
  float s = 0.f;
  for (int i = threadIdx.x; i < HW_; i += 256) s += p[i];
  __shared__ float red[256];
  red[threadIdx.x] = s;
  __syncthreads();
  for (int st = 128; st > 0; st >>= 1) {
    if (threadIdx.x < st) red[threadIdx.x] += red[threadIdx.x + st];
    __syncthreads();
  }
  if (threadIdx.x == 0) ctx[blc] = red[0] * (1.0f / (float)HW_);
}

// ---------------- K1: MLP + lambda/gamma ----------------
__global__ void k_mlp(const float* __restrict__ ctx, const float* __restrict__ w1,
                      const float* __restrict__ b1, const float* __restrict__ w2,
                      const float* __restrict__ b2, const float* __restrict__ fs_p,
                      const float* __restrict__ nu_log,
                      const float* __restrict__ th_log, float* __restrict__ lam_r,
                      float* __restrict__ lam_i, float* __restrict__ gam) {
  int bl = blockIdx.x;
  __shared__ float sc[C_];
  __shared__ float sh[MH_];
  int tid = threadIdx.x;
  if (tid < C_) sc[tid] = ctx[bl * C_ + tid];
  __syncthreads();
  if (tid < MH_) {
    float a = b1[tid];
    for (int c = 0; c < C_; ++c) a += sc[c] * w1[c * MH_ + tid];
    sh[tid] = tanhf(a);
  }
  __syncthreads();
  float fs = fs_p[0];
  for (int j = tid; j < C_ * R_; j += 256) {
    float d0 = b2[j], d1 = b2[C_ * R_ + j];
    for (int m = 0; m < MH_; ++m) {
      float hv = sh[m];
      d0 += hv * w2[m * (2 * C_ * R_) + j];
      d1 += hv * w2[m * (2 * C_ * R_) + C_ * R_ + j];
    }
    float nu = expf(nu_log[j] + fs * d0);
    float th = expf(th_log[j] + fs * d1);
    float mag = expf(-nu);
    float s, co;
    __sincosf(th, &s, &co);
    size_t o = (size_t)bl * C_ * R_ + j;
    lam_r[o] = mag * co;
    lam_i[o] = mag * s;
    gam[o] = sqrtf(fmaxf(0.f, 1.f - expf(-2.f * nu)));
  }
}

// ---------------- K2: DFT-folded U/V factors ----------------
__global__ void k_factU(const float* __restrict__ Ur, const float* __restrict__ Ui,
                        float* __restrict__ Uh_r, float* __restrict__ Uh_i,
                        float* __restrict__ Ut_r, float* __restrict__ Ut_i) {
  int idx = blockIdx.x * blockDim.x + threadIdx.x;  // (c,p,r)
  if (idx >= C_ * H_ * R_) return;
  int r = idx % R_;
  int p = (idx / R_) % H_;
  int c = idx / (R_ * H_);
  const float* br = Ur + (size_t)c * H_ * R_ + r;
  const float* bi = Ui + (size_t)c * H_ * R_ + r;
  float hr = 0, hi = 0, tr = 0, ti = 0;
  float w0 = -2.0f * PI_F * (float)p / (float)H_;
  for (int h = 0; h < H_; ++h) {
    float s, co;
    __sincosf(w0 * (float)h, &s, &co);
    float ur = br[(size_t)h * R_], ui = bi[(size_t)h * R_];
    hr += co * ur - s * ui;
    hi += co * ui + s * ur;
    tr += co * ur + s * ui;
    ti += co * ui - s * ur;
  }
  Uh_r[idx] = hr;
  Uh_i[idx] = hi;
  Ut_r[idx] = tr * (1.0f / (float)H_);
  Ut_i[idx] = ti * (1.0f / (float)H_);
}

__global__ void k_factV(const float* __restrict__ Vr, const float* __restrict__ Vi,
                        float* __restrict__ Vh_r, float* __restrict__ Vh_i,
                        float* __restrict__ Vt_r, float* __restrict__ Vt_i) {
  int idx = blockIdx.x * blockDim.x + threadIdx.x;  // (c,q,r)
  if (idx >= C_ * W_ * R_) return;
  int r = idx % R_;
  int q = (idx / R_) % W_;
  int c = idx / (R_ * W_);
  const float* br = Vr + (size_t)c * W_ * R_ + r;
  const float* bi = Vi + (size_t)c * W_ * R_ + r;
  float hr = 0, hi = 0, tr = 0, ti = 0;
  float w0 = -2.0f * PI_F * (float)q / (float)W_;
  for (int w = 0; w < W_; ++w) {
    float s, co;
    __sincosf(w0 * (float)w, &s, &co);
    float ur = br[(size_t)w * R_], ui = bi[(size_t)w * R_];
    hr += co * ur - s * ui;
    hi += co * ui + s * ur;
    tr += co * ur + s * ui;
    ti += co * ui - s * ur;
  }
  Vh_r[idx] = hr;
  Vh_i[idx] = hi;
  Vt_r[idx] = tr * (1.0f / (float)W_);
  Vt_i[idx] = ti * (1.0f / (float)W_);
}

// ---------------- K2b: fold fuse_w into conv weights -> bf16, K padded 144->160 ----
constexpr int KC = 16;                 // channels per k-chunk
constexpr int CHUNKS = (2 * C_) / KC;  // 8
constexpr int KPAD = 160;              // 16*9=144 padded to multiple of 32

__global__ void k_fusew(const float* __restrict__ fuse_w,
                        const float* __restrict__ fuse_b,
                        const float* __restrict__ cr_w, const float* __restrict__ cr_b,
                        const float* __restrict__ ci_w, const float* __restrict__ ci_b,
                        unsigned short* __restrict__ A2bf, float* __restrict__ fb) {
  int idx = blockIdx.x * blockDim.x + threadIdx.x;  // o*(8*160) + kc*160 + pos
  if (idx < C_ * CHUNKS * KPAD) {
    int pos = idx % KPAD;
    int kc = (idx / KPAD) % CHUNKS;
    int o = idx / (KPAD * CHUNKS);
    float s = 0.f;
    if (pos < KC * 9) {
      int k = kc * KC + pos / 9;
      int t = pos % 9;
      if (k < C_) {
        for (int cp = 0; cp < C_; ++cp)
          s += fuse_w[o * 2 * C_ + cp] * cr_w[((size_t)cp * C_ + k) * 9 + t];
      } else {
        int kk = k - C_;
        for (int cp = 0; cp < C_; ++cp)
          s += fuse_w[o * 2 * C_ + C_ + cp] * ci_w[((size_t)cp * C_ + kk) * 9 + t];
      }
    }
    A2bf[idx] = f2bf(s);
  }
  if (idx < C_) {
    float s = fuse_b[idx];
    for (int cp = 0; cp < C_; ++cp)
      s += fuse_w[idx * 2 * C_ + cp] * cr_b[cp] +
           fuse_w[idx * 2 * C_ + C_ + cp] * ci_b[cp];
    fb[idx] = s;
  }
}

// ---------------- tiny f32 -> bf16 weight converter ----------------
__global__ void k_tobf16(const float* __restrict__ a, unsigned short* __restrict__ o,
                         int n) {
  int i = blockIdx.x * blockDim.x + threadIdx.x;
  if (i < n) o[i] = f2bf(a[i]);
}

// ---------------- K3: rank-R encode u = x . Vhat . Uhat ----------------
__global__ void k_encode(const float* __restrict__ x, const float* __restrict__ Vh_r,
                         const float* __restrict__ Vh_i,
                         const float* __restrict__ Uh_r,
                         const float* __restrict__ Uh_i, float* __restrict__ u_r,
                         float* __restrict__ u_i) {
  int blc = blockIdx.x;
  int c = blc % C_;
  __shared__ float sx[HW_];  // 32KB
  __shared__ float redr[256], redi[256];
  const float* xp = x + (size_t)blc * HW_;
  for (int i = threadIdx.x; i < HW_; i += 256) sx[i] = xp[i];
  __syncthreads();
  int r = threadIdx.x & 31, grp = threadIdx.x >> 5;
  const float* vr = Vh_r + (size_t)c * W_ * R_ + r;
  const float* vi = Vh_i + (size_t)c * W_ * R_ + r;
  float accr = 0.f, acci = 0.f;
  for (int pp = 0; pp < 8; ++pp) {
    int p = grp * 8 + pp;
    float sr = 0.f, si = 0.f;
    for (int q = 0; q < W_; ++q) {
      float xv = sx[p * W_ + q];
      sr += xv * vr[(size_t)q * R_];
      si += xv * vi[(size_t)q * R_];
    }
    float ur = Uh_r[((size_t)c * H_ + p) * R_ + r];
    float ui = Uh_i[((size_t)c * H_ + p) * R_ + r];
    accr += sr * ur - si * ui;
    acci += sr * ui + si * ur;
  }
  redr[threadIdx.x] = accr;
  redi[threadIdx.x] = acci;
  __syncthreads();
  if (grp == 0) {
    for (int g = 1; g < 8; ++g) {
      accr += redr[g * 32 + r];
      acci += redi[g * 32 + r];
    }
    u_r[(size_t)blc * R_ + r] = accr;
    u_i[(size_t)blc * R_ + r] = acci;
  }
}

// ---------------- K4: sequential LRU scan over L ----------------
__global__ void k_scan(const float* __restrict__ lam_r, const float* __restrict__ lam_i,
                       const float* __restrict__ gam, const float* __restrict__ u_r,
                       const float* __restrict__ u_i, float* __restrict__ hs_r,
                       float* __restrict__ hs_i) {
  int idx = blockIdx.x * blockDim.x + threadIdx.x;
  if (idx >= B_ * C_ * R_) return;
  int b = idx / (C_ * R_);
  int cr = idx % (C_ * R_);
  float hr = 0.f, hi = 0.f;
  for (int l = 0; l < L_; ++l) {
    size_t o = ((size_t)(b * L_ + l)) * C_ * R_ + cr;
    float lr = lam_r[o], li = lam_i[o], g = gam[o];
    float nr = lr * hr - li * hi + g * u_r[o];
    float ni = lr * hi + li * hr + g * u_i[o];
    hr = nr;
    hi = ni;
    hs_r[o] = hr;
    hs_i[o] = hi;
  }
}

// ---------------- K5: low-mode DFT of x ----------------
__global__ void k_xflow(const float* __restrict__ x, float* __restrict__ xfl_r,
                        float* __restrict__ xfl_i) {
  int blc = blockIdx.x;
  __shared__ float sx[HW_];
  __shared__ float twr[M2_ * W_], twi[M2_ * W_];
  __shared__ float t1r[H_ * M2_], t1i[H_ * M2_];
  const float* xp = x + (size_t)blc * HW_;
  for (int i = threadIdx.x; i < HW_; i += 256) sx[i] = xp[i];
  for (int i = threadIdx.x; i < M2_ * W_; i += 256) {
    int w = i / W_, q = i % W_;
    float s, c;
    __sincosf(-2.0f * PI_F * (float)(w * q) / (float)W_, &s, &c);
    twr[i] = c;
    twi[i] = s;
  }
  __syncthreads();
  for (int i = threadIdx.x; i < H_ * M2_; i += 256) {
    int p = i / M2_, w = i % M2_;
    float ar = 0.f, ai = 0.f;
    for (int q = 0; q < W_; ++q) {
      float xv = sx[p * W_ + q];
      ar += xv * twr[w * W_ + q];
      ai += xv * twi[w * W_ + q];
    }
    t1r[i] = ar;
    t1i[i] = ai;
  }
  __syncthreads();
  if (threadIdx.x < HM_ * M2_) {
    int hm = threadIdx.x / M2_, w = threadIdx.x % M2_;
    int h = (hm < 8) ? hm : (48 + hm);
    float ar = 0.f, ai = 0.f;
    for (int p = 0; p < H_; ++p) {
      float s, c;
      __sincosf(-2.0f * PI_F * (float)(h * p) / (float)H_, &s, &c);
      float tr = t1r[p * M2_ + w], ti = t1i[p * M2_ + w];
      ar += c * tr - s * ti;
      ai += c * ti + s * tr;
    }
    size_t o = (size_t)blc * HM_ * M2_ + threadIdx.x;
    xfl_r[o] = ar;
    xfl_i[o] = ai;
  }
}

// ---------------- K6: spectral channel mix at low modes ----------------
__global__ void k_specmix(const float* __restrict__ xfl_r,
                          const float* __restrict__ xfl_i,
                          const float* __restrict__ swr1, const float* __restrict__ swi1,
                          const float* __restrict__ swr2, const float* __restrict__ swi2,
                          float* __restrict__ spl_r, float* __restrict__ spl_i) {
  int bl = blockIdx.x;
  __shared__ float sxr[C_ * HM_ * M2_];  // 32KB
  const float* pr = xfl_r + (size_t)bl * C_ * HM_ * M2_;
  const float* pi = xfl_i + (size_t)bl * C_ * HM_ * M2_;
  for (int i = threadIdx.x; i < C_ * HM_ * M2_; i += 256) sxr[i] = pr[i];
  __syncthreads();
  for (int i = threadIdx.x; i < C_ * HM_ * M2_; i += 256) {
    int w = i % M2_;
    int hm = (i / M2_) % HM_;
    int o = i / (M2_ * HM_);
    const float* wr;
    const float* wi;
    int xx;
    if (hm < 8) {
      wr = swr1; wi = swi1; xx = hm;
    } else {
      wr = swr2; wi = swi2; xx = hm - 8;
    }
    float ar = 0.f, ai = 0.f;
    for (int c = 0; c < C_; ++c) {
      float xr = sxr[(c * HM_ + hm) * M2_ + w];
      float xi = pi[(c * HM_ + hm) * M2_ + w];
      size_t widx = (((size_t)o * C_ + c) * M1_ + xx) * M2_ + w;
      float wrv = wr[widx], wiv = wi[widx];
      ar += wrv * xr - wiv * xi;
      ai += wrv * xi + wiv * xr;
    }
    spl_r[(size_t)bl * C_ * HM_ * M2_ + i] = ar;
    spl_i[(size_t)bl * C_ * HM_ * M2_ + i] = ai;
  }
}

// ---------------- K8: decode yd = (hs*Ut) x VtT  (complex bf16 WMMA) ----------------
__global__ void k_decode(const float* __restrict__ hs_r, const float* __restrict__ hs_i,
                         const float* __restrict__ Ut_r, const float* __restrict__ Ut_i,
                         const float* __restrict__ Vt_r, const float* __restrict__ Vt_i,
                         float* __restrict__ yd_r, float* __restrict__ yd_i) {
  int blc = blockIdx.x;
  int c = blc % C_;
  __shared__ alignas(16) unsigned short mbr[H_ * R_], mbi[H_ * R_];  // 4KB each
  __shared__ alignas(16) unsigned short vbr[W_ * R_], vbi[W_ * R_];  // 8KB each
  for (int i = threadIdx.x; i < H_ * R_; i += 256) {
    int p = i / R_, r = i % R_;
    float hr = hs_r[(size_t)blc * R_ + r], hi = hs_i[(size_t)blc * R_ + r];
    float ur = Ut_r[((size_t)c * H_ + p) * R_ + r];
    float ui = Ut_i[((size_t)c * H_ + p) * R_ + r];
    mbr[i] = f2bf(hr * ur - hi * ui);
    mbi[i] = f2bf(hr * ui + hi * ur);
  }
  for (int i = threadIdx.x; i < W_ * R_; i += 256) {
    vbr[i] = f2bf(Vt_r[(size_t)c * W_ * R_ + i]);
    vbi[i] = f2bf(Vt_i[(size_t)c * W_ * R_ + i]);
  }
  __syncthreads();
  int wave = threadIdx.x >> 5, lane = threadIdx.x & 31;
#pragma unroll
  for (int t = 0; t < 4; ++t) {
    int tile = wave * 4 + t;
    int pt = tile >> 3, qt = tile & 7;  // M=64 (p), N=128 (q)
    v8f aRR = v8f_zero(), aII = v8f_zero(), aRI = v8f_zero(), aIR = v8f_zero();
    const unsigned short* Ar = mbr + pt * 16 * R_;
    const unsigned short* Ai = mbi + pt * 16 * R_;
    const unsigned short* Br = vbr + qt * 16 * R_;  // Vt is [q][r] = [N][K]
    const unsigned short* Bi = vbi + qt * 16 * R_;
    v16bf ar = load_a_bf(Ar, R_, 0, lane);
    v16bf ai = load_a_bf(Ai, R_, 0, lane);
    v16bf br = load_bt_bf(Br, R_, 0, lane);
    v16bf bi = load_bt_bf(Bi, R_, 0, lane);
    aRR = wmma_bf(ar, br, aRR);
    aII = wmma_bf(ai, bi, aII);
    aRI = wmma_bf(ar, bi, aRI);
    aIR = wmma_bf(ai, br, aIR);
    int n = lane & 15, mh = (lane >> 4) * 8;
    float* dr = yd_r + (size_t)blc * HW_ + pt * 16 * W_ + qt * 16;
    float* di = yd_i + (size_t)blc * HW_ + pt * 16 * W_ + qt * 16;
#pragma unroll
    for (int vg = 0; vg < 8; ++vg) {
      int mrow = mh + vg;
      dr[(size_t)mrow * W_ + n] = aRR[vg] - aII[vg];
      di[(size_t)mrow * W_ + n] = aRI[vg] + aIR[vg];
    }
  }
}

// ---------------- K10: channel projection (complex bf16 WMMA) ----------------
__global__ void k_proj(const float* __restrict__ yd_r, const float* __restrict__ yd_i,
                       const unsigned short* __restrict__ Pbr,
                       const unsigned short* __restrict__ Pbi, float* __restrict__ y_r,
                       float* __restrict__ y_i) {
  int bl = blockIdx.x >> 6;  // 64 chunks of 128 pixels
  int chunk = blockIdx.x & 63;
  int pix0 = chunk * 128;
  __shared__ alignas(16) unsigned short sBtr[128 * C_], sBti[128 * C_];  // 16KB each
  for (int i = threadIdx.x; i < C_ * 128; i += 256) {
    int cc = i >> 7, j = i & 127;  // coalesced read, transposed [pix][c] write
    size_t src = ((size_t)(bl * C_ + cc)) * HW_ + pix0 + j;
    sBtr[j * C_ + cc] = f2bf(yd_r[src]);
    sBti[j * C_ + cc] = f2bf(yd_i[src]);
  }
  __syncthreads();
  int wave = threadIdx.x >> 5, lane = threadIdx.x & 31;
#pragma unroll
  for (int t = 0; t < 4; ++t) {
    int tile = wave * 4 + t;
    int mt = tile >> 3, nt = tile & 7;  // M=64 (o), N=128 (pix)
    v8f aRR = v8f_zero(), aII = v8f_zero(), aRI = v8f_zero(), aIR = v8f_zero();
    const unsigned short* Ar = Pbr + mt * 16 * C_;  // bf16 weights from global
    const unsigned short* Ai = Pbi + mt * 16 * C_;
    const unsigned short* Br = sBtr + nt * 16 * C_;
    const unsigned short* Bi = sBti + nt * 16 * C_;
#pragma unroll
    for (int kt = 0; kt < C_ / 32; ++kt) {
      v16bf ar = load_a_bf(Ar, C_, kt, lane);
      v16bf ai = load_a_bf(Ai, C_, kt, lane);
      v16bf br = load_bt_bf(Br, C_, kt, lane);
      v16bf bi = load_bt_bf(Bi, C_, kt, lane);
      aRR = wmma_bf(ar, br, aRR);
      aII = wmma_bf(ai, bi, aII);
      aRI = wmma_bf(ar, bi, aRI);
      aIR = wmma_bf(ai, br, aIR);
    }
    int n = lane & 15, mh = (lane >> 4) * 8;
#pragma unroll
    for (int vg = 0; vg < 8; ++vg) {
      int o = mt * 16 + mh + vg;
      size_t dst = ((size_t)(bl * C_ + o)) * HW_ + pix0 + nt * 16 + n;
      y_r[dst] = aRR[vg] - aII[vg];
      y_i[dst] = aRI[vg] + aIR[vg];
    }
  }
}

// ---------------- K7: inverse transform of low-mode prior, accumulate ----------------
__global__ void k_spifft(const float* __restrict__ spl_r,
                         const float* __restrict__ spl_i,
                         const float* __restrict__ pbr, const float* __restrict__ pbi,
                         float* __restrict__ y_r, float* __restrict__ y_i) {
  int blo = blockIdx.x;
  int o = blo % C_;
  __shared__ float ssr[HM_ * M2_], ssi[HM_ * M2_];
  __shared__ float tr_[HM_ * W_], ti_[HM_ * W_];
  __shared__ float hwr[HM_ * H_], hwi[HM_ * H_];
  for (int i = threadIdx.x; i < HM_ * M2_; i += 256) {
    ssr[i] = spl_r[(size_t)blo * HM_ * M2_ + i];
    ssi[i] = spl_i[(size_t)blo * HM_ * M2_ + i];
  }
  for (int i = threadIdx.x; i < HM_ * H_; i += 256) {
    int hm = i / H_, pp = i % H_;
    int h = (hm < 8) ? hm : (48 + hm);
    float s, c;
    __sincosf(2.0f * PI_F * (float)(h * pp) / (float)H_, &s, &c);
    hwr[i] = c;
    hwi[i] = s;
  }
  __syncthreads();
  for (int i = threadIdx.x; i < HM_ * W_; i += 256) {
    int hm = i >> 7, q = i & 127;
    float ar = 0.f, ai = 0.f;
    for (int w = 0; w < M2_; ++w) {
      float s, c;
      __sincosf(2.0f * PI_F * (float)(w * q) / (float)W_, &s, &c);
      float xr = ssr[hm * M2_ + w], xi = ssi[hm * M2_ + w];
      ar += c * xr - s * xi;
      ai += c * xi + s * xr;
    }
    tr_[i] = ar;
    ti_[i] = ai;
  }
  __syncthreads();
  float* dr = y_r + (size_t)blo * HW_;
  float* di = y_i + (size_t)blo * HW_;
  for (int i = threadIdx.x; i < HW_; i += 256) {
    int p = i >> 7, q = i & 127;
    float ar = 0.f, ai = 0.f;
    for (int hm = 0; hm < HM_; ++hm) {
      float c = hwr[hm * H_ + p], s = hwi[hm * H_ + p];
      float xr = tr_[hm * W_ + q], xi = ti_[hm * W_ + q];
      ar += c * xr - s * xi;
      ai += c * xi + s * xr;
    }
    float addr = ar * (1.0f / (float)HW_), addi = ai * (1.0f / (float)HW_);
    if (i == 0) {  // projb -> delta at (0,0)
      addr += pbr[o];
      addi += pbi[o];
    }
    dr[i] += addr;
    di[i] += addi;
  }
}

// ---------------- K11: gate pre-activation GEMM (bf16 WMMA) ----------------
__global__ void k_gate(const float* __restrict__ x, const unsigned short* __restrict__ Gbf,
                       const float* __restrict__ Gb, float* __restrict__ gl) {
  int bl = blockIdx.x >> 6;
  int chunk = blockIdx.x & 63;
  int pix0 = chunk * 128;
  __shared__ alignas(16) unsigned short sXt[128 * C_];  // 16KB, transposed [pix][c]
  __shared__ float sGb[C_];
  for (int i = threadIdx.x; i < C_ * 128; i += 256) {
    int cc = i >> 7, j = i & 127;
    sXt[j * C_ + cc] = f2bf(x[((size_t)(bl * C_ + cc)) * HW_ + pix0 + j]);
  }
  if (threadIdx.x < C_) sGb[threadIdx.x] = Gb[threadIdx.x];
  __syncthreads();
  int wave = threadIdx.x >> 5, lane = threadIdx.x & 31;
#pragma unroll
  for (int t = 0; t < 4; ++t) {
    int tile = wave * 4 + t;
    int mt = tile >> 3, nt = tile & 7;  // M=64, N=128
    v8f acc = v8f_zero();
    const unsigned short* Ar = Gbf + mt * 16 * C_;
    const unsigned short* Br = sXt + nt * 16 * C_;
#pragma unroll
    for (int kt = 0; kt < C_ / 32; ++kt) {
      v16bf a = load_a_bf(Ar, C_, kt, lane);
      v16bf b = load_bt_bf(Br, C_, kt, lane);
      acc = wmma_bf(a, b, acc);
    }
    int n = lane & 15, mh = (lane >> 4) * 8;
#pragma unroll
    for (int vg = 0; vg < 8; ++vg) {
      int o = mt * 16 + mh + vg;
      gl[((size_t)(bl * C_ + o)) * HW_ + pix0 + nt * 16 + n] = acc[vg] + sGb[o];
    }
  }
}

// ---------------- K12: fused 3x3 conv (implicit GEMM, bf16 WMMA) ----------------
__global__ void k_conv(const float* __restrict__ y_r, const float* __restrict__ y_i,
                       const unsigned short* __restrict__ A2bf,
                       const float* __restrict__ fb, float* __restrict__ out) {
  int blk = blockIdx.x;  // bl*(H*2) + p*2 + qh
  int qh = blk & 1;
  int p = (blk >> 1) & (H_ - 1);
  int bl = blk >> 7;
  int q0 = qh * 64;
  __shared__ alignas(16) unsigned short sA[C_ * KPAD];   // 20KB, [o][K]
  __shared__ alignas(16) unsigned short sZt[64 * KPAD];  // 20KB, im2col [pix][K]
  int wave = threadIdx.x >> 5, lane = threadIdx.x & 31;
  int tid = threadIdx.x;
  v8f acc0 = v8f_zero(), acc1 = v8f_zero();
  for (int kc = 0; kc < CHUNKS; ++kc) {
    __syncthreads();
    // stage weights (contiguous bf16 copy as dwords)
    for (int i = tid; i < C_ * KPAD / 2; i += 256) {
      int o = i / (KPAD / 2), pos2 = i % (KPAD / 2);
      ((unsigned int*)sA)[i] =
          ((const unsigned int*)A2bf)[(size_t)o * (CHUNKS * KPAD / 2) +
                                      kc * (KPAD / 2) + pos2];
    }
    // stage im2col tile, transposed [pix][K], zero-padded K 144..159
    for (int i = tid; i < 64 * KPAD; i += 256) {
      int pos = i % KPAD;
      int n = i / KPAD;
      float v = 0.f;
      if (pos < KC * 9) {
        int zc = pos / 9, rem = pos % 9;
        int i3 = rem / 3, j = rem % 3;
        int ch = kc * KC + zc;
        const float* plane = (ch < C_)
                                 ? (y_r + ((size_t)(bl * C_ + ch)) * HW_)
                                 : (y_i + ((size_t)(bl * C_ + (ch - C_))) * HW_);
        int sp = p + i3 - 1;
        int sq = q0 + n + j - 1;
        if (sp >= 0 && sp < H_ && sq >= 0 && sq < W_) v = plane[sp * W_ + sq];
      }
      sZt[i] = f2bf(v);
    }
    __syncthreads();
#pragma unroll
    for (int t = 0; t < 2; ++t) {
      int id = wave * 2 + t;
      int mt = id >> 2, nt = id & 3;  // M=64, N=64
      const unsigned short* Ar = sA + mt * 16 * KPAD;
      const unsigned short* Br = sZt + nt * 16 * KPAD;
      v8f acc = (t == 0) ? acc0 : acc1;
#pragma unroll
      for (int kt = 0; kt < KPAD / 32; ++kt) {
        v16bf a = load_a_bf(Ar, KPAD, kt, lane);
        v16bf b = load_bt_bf(Br, KPAD, kt, lane);
        acc = wmma_bf(a, b, acc);
      }
      if (t == 0) acc0 = acc; else acc1 = acc;
    }
  }
  int n = lane & 15, mh = (lane >> 4) * 8;
#pragma unroll
  for (int t = 0; t < 2; ++t) {
    int id = wave * 2 + t;
    int mt = id >> 2, nt = id & 3;
    v8f acc = (t == 0) ? acc0 : acc1;
#pragma unroll
    for (int vg = 0; vg < 8; ++vg) {
      int o = mt * 16 + mh + vg;
      out[((size_t)(bl * C_ + o)) * HW_ + p * W_ + q0 + nt * 16 + n] =
          acc[vg] + fb[o];
    }
  }
}

// ---------------- K13: LayerNorm statistics ----------------
__global__ void k_lnstat(const float* __restrict__ out, float* __restrict__ mu,
                         float* __restrict__ var) {
  int blo = blockIdx.x;
  const float* p = out + (size_t)blo * HW_;
  float s = 0.f, s2 = 0.f;
  for (int i = threadIdx.x; i < HW_; i += 256) {
    float v = p[i];
    s += v;
    s2 += v * v;
  }
  __shared__ float r1[256], r2[256];
  r1[threadIdx.x] = s;
  r2[threadIdx.x] = s2;
  __syncthreads();
  for (int st = 128; st > 0; st >>= 1) {
    if (threadIdx.x < st) {
      r1[threadIdx.x] += r1[threadIdx.x + st];
      r2[threadIdx.x] += r2[threadIdx.x + st];
    }
    __syncthreads();
  }
  if (threadIdx.x == 0) {
    float m = r1[0] * (1.0f / (float)HW_);
    mu[blo] = m;
    var[blo] = r2[0] * (1.0f / (float)HW_) - m * m;
  }
}

// ---------------- K14: final LN + gate + residual ----------------
__global__ void k_final(const float* __restrict__ x, const float* __restrict__ gl,
                        const float* __restrict__ mu, const float* __restrict__ var,
                        const float* __restrict__ ln_w, const float* __restrict__ ln_b,
                        float* __restrict__ out) {
  size_t idx = (size_t)blockIdx.x * blockDim.x + threadIdx.x;
  size_t total = (size_t)BL_ * C_ * HW_;
  if (idx >= total) return;
  int pix = (int)(idx % HW_);
  size_t blo = idx / HW_;
  float m = mu[blo], v = var[blo];
  float o = out[idx];
  float norm = (o - m) * rsqrtf(v + 1e-5f) * ln_w[pix] + ln_b[pix];
  float g = 1.f / (1.f + expf(-gl[idx]));
  out[idx] = x[idx] + g * norm;
}

// ---------------- host launcher ----------------
extern "C" void kernel_launch(void* const* d_in, const int* in_sizes, int n_in,
                              void* d_out, int out_size, void* d_ws, size_t ws_size,
                              hipStream_t stream) {
  (void)in_sizes; (void)n_in; (void)out_size; (void)ws_size;
  const float* x        = (const float*)d_in[0];
  const float* nu_log   = (const float*)d_in[1];
  const float* th_log   = (const float*)d_in[2];
  const float* mlp_w1   = (const float*)d_in[3];
  const float* mlp_b1   = (const float*)d_in[4];
  const float* mlp_w2   = (const float*)d_in[5];
  const float* mlp_b2   = (const float*)d_in[6];
  const float* fscale   = (const float*)d_in[7];
  const float* U_r      = (const float*)d_in[8];
  const float* U_i      = (const float*)d_in[9];
  const float* V_r      = (const float*)d_in[10];
  const float* V_i      = (const float*)d_in[11];
  const float* projW_r  = (const float*)d_in[12];
  const float* projW_i  = (const float*)d_in[13];
  const float* projb_r  = (const float*)d_in[14];
  const float* projb_i  = (const float*)d_in[15];
  const float* swr1     = (const float*)d_in[16];
  const float* swi1     = (const float*)d_in[17];
  const float* swr2     = (const float*)d_in[18];
  const float* swi2     = (const float*)d_in[19];
  const float* convr_w  = (const float*)d_in[20];
  const float* convr_b  = (const float*)d_in[21];
  const float* convi_w  = (const float*)d_in[22];
  const float* convi_b  = (const float*)d_in[23];
  const float* fuse_w   = (const float*)d_in[24];
  const float* fuse_b   = (const float*)d_in[25];
  const float* gate_w   = (const float*)d_in[26];
  const float* gate_b   = (const float*)d_in[27];
  const float* ln_w     = (const float*)d_in[28];
  const float* ln_b     = (const float*)d_in[29];
  float* out = (float*)d_out;

  // workspace carve-up (float units)
  float* wsf = (float*)d_ws;
  size_t off = 0;
  auto take = [&](size_t n) { float* p = wsf + off; off += n; return p; };
  float* ctx   = take(BL_ * C_);
  float* lam_r = take((size_t)BL_ * C_ * R_);
  float* lam_i = take((size_t)BL_ * C_ * R_);
  float* gam   = take((size_t)BL_ * C_ * R_);
  float* u_r   = take((size_t)BL_ * C_ * R_);
  float* u_i   = take((size_t)BL_ * C_ * R_);
  float* hs_r  = take((size_t)BL_ * C_ * R_);
  float* hs_i  = take((size_t)BL_ * C_ * R_);
  float* Uh_r  = take((size_t)C_ * H_ * R_);
  float* Uh_i  = take((size_t)C_ * H_ * R_);
  float* Ut_r  = take((size_t)C_ * H_ * R_);
  float* Ut_i  = take((size_t)C_ * H_ * R_);
  float* Vh_r  = take((size_t)C_ * W_ * R_);
  float* Vh_i  = take((size_t)C_ * W_ * R_);
  float* Vt_r  = take((size_t)C_ * W_ * R_);
  float* Vt_i  = take((size_t)C_ * W_ * R_);
  float* xfl_r = take((size_t)BL_ * C_ * HM_ * M2_);
  float* xfl_i = take((size_t)BL_ * C_ * HM_ * M2_);
  float* spl_r = take((size_t)BL_ * C_ * HM_ * M2_);
  float* spl_i = take((size_t)BL_ * C_ * HM_ * M2_);
  unsigned short* A2bf = (unsigned short*)take((size_t)C_ * CHUNKS * KPAD / 2);
  unsigned short* Pbfr = (unsigned short*)take((size_t)C_ * C_ / 2);
  unsigned short* Pbfi = (unsigned short*)take((size_t)C_ * C_ / 2);
  unsigned short* Gbf  = (unsigned short*)take((size_t)C_ * C_ / 2);
  float* fb    = take(C_);
  float* mu    = take(BL_ * C_);
  float* var   = take(BL_ * C_);
  float* yd_r  = take((size_t)BL_ * C_ * HW_);
  float* yd_i  = take((size_t)BL_ * C_ * HW_);
  float* y_r   = take((size_t)BL_ * C_ * HW_);
  float* y_i   = take((size_t)BL_ * C_ * HW_);
  float* gl    = yd_r;  // reused after k_proj consumed yd

  dim3 blk(256);
  k_ctx<<<dim3(BL_ * C_), blk, 0, stream>>>(x, ctx);
  k_mlp<<<dim3(BL_), blk, 0, stream>>>(ctx, mlp_w1, mlp_b1, mlp_w2, mlp_b2, fscale,
                                       nu_log, th_log, lam_r, lam_i, gam);
  k_factU<<<dim3((C_ * H_ * R_ + 255) / 256), blk, 0, stream>>>(U_r, U_i, Uh_r, Uh_i,
                                                               Ut_r, Ut_i);
  k_factV<<<dim3((C_ * W_ * R_ + 255) / 256), blk, 0, stream>>>(V_r, V_i, Vh_r, Vh_i,
                                                               Vt_r, Vt_i);
  k_fusew<<<dim3((C_ * CHUNKS * KPAD + 255) / 256), blk, 0, stream>>>(
      fuse_w, fuse_b, convr_w, convr_b, convi_w, convi_b, A2bf, fb);
  k_tobf16<<<dim3((C_ * C_ + 255) / 256), blk, 0, stream>>>(projW_r, Pbfr, C_ * C_);
  k_tobf16<<<dim3((C_ * C_ + 255) / 256), blk, 0, stream>>>(projW_i, Pbfi, C_ * C_);
  k_tobf16<<<dim3((C_ * C_ + 255) / 256), blk, 0, stream>>>(gate_w, Gbf, C_ * C_);
  k_encode<<<dim3(BL_ * C_), blk, 0, stream>>>(x, Vh_r, Vh_i, Uh_r, Uh_i, u_r, u_i);
  k_scan<<<dim3((B_ * C_ * R_ + 255) / 256), blk, 0, stream>>>(lam_r, lam_i, gam, u_r,
                                                              u_i, hs_r, hs_i);
  k_xflow<<<dim3(BL_ * C_), blk, 0, stream>>>(x, xfl_r, xfl_i);
  k_specmix<<<dim3(BL_), blk, 0, stream>>>(xfl_r, xfl_i, swr1, swi1, swr2, swi2,
                                           spl_r, spl_i);
  k_decode<<<dim3(BL_ * C_), blk, 0, stream>>>(hs_r, hs_i, Ut_r, Ut_i, Vt_r, Vt_i,
                                               yd_r, yd_i);
  k_proj<<<dim3(BL_ * 64), blk, 0, stream>>>(yd_r, yd_i, Pbfr, Pbfi, y_r, y_i);
  k_spifft<<<dim3(BL_ * C_), blk, 0, stream>>>(spl_r, spl_i, projb_r, projb_i, y_r,
                                               y_i);
  k_gate<<<dim3(BL_ * 64), blk, 0, stream>>>(x, Gbf, gate_b, gl);
  k_conv<<<dim3(BL_ * H_ * 2), blk, 0, stream>>>(y_r, y_i, A2bf, fb, out);
  k_lnstat<<<dim3(BL_ * C_), blk, 0, stream>>>(out, mu, var);
  size_t total = (size_t)BL_ * C_ * HW_;
  k_final<<<dim3((unsigned)((total + 255) / 256)), blk, 0, stream>>>(x, gl, mu, var,
                                                                    ln_w, ln_b, out);
}
